// Decoder_32418413150696
// MI455X (gfx1250) — compile-verified, matrix operand order checked by reading, not run on hardware
//
#include <hip/hip_runtime.h>
#include <hip/hip_bf16.h>
#include <math.h>

// ---------------------------------------------------------------------------
// Problem constants (from reference): S=64, N=256, H=1024, E=256, V=128, DIR=2
// ---------------------------------------------------------------------------
#define S_LEN 64
#define N_BATCH 256
#define H_DIM 1024
#define E_DIM 256
#define V_DIM 128
#define DH (2 * H_DIM)          // 2048
#define RNN_IN (DH + E_DIM)     // 2304
#define G3H (3 * H_DIM)         // 3072

typedef __attribute__((ext_vector_type(16))) __bf16 v16bf;
typedef __attribute__((ext_vector_type(8)))  __bf16 v8bf;
typedef __attribute__((ext_vector_type(4)))  __bf16 v4bf;
typedef __attribute__((ext_vector_type(8)))  float  v8f;

__device__ __forceinline__ float sigmoidf_(float x) {
    return 1.0f / (1.0f + expf(-x));
}

// ---------------------------------------------------------------------------
// bf16-WMMA GEMM:  C[M,Nout] = A[M,K] * W[Nout,K]^T (+ bias)
//   256 threads = 8 waves, block tile 128x128, wave grid 4(M)x2(N),
//   wave tile 32x64 (2 A-frags x 4 B-frags = 8 v_wmma per K-step),
//   K-step 32, double-buffered LDS, global->reg prefetch across the WMMAs,
//   fp32 global data converted to bf16 via native HW converts on the fly.
// mode 0: Cout[m*Nout+col] = acc + bias[col]
// mode 1: attention-energy epilogue:
//           val = tanh(acc + bias[col] + wbuf[(m&nMask)*Hdim + col]) * attn_w[col]
//           row-sums via shfl_xor + small LDS combine ->
//           partials[m*gridDim.x + blockIdx.x]   (deterministic, no atomics)
// LDS map (bytes): [0,8K) A buf0 | [8K,16K) A buf1 | [16K,24K) B buf0
//                  [24K,32K) B buf1 | [32K,33K) reduce buffer
// ---------------------------------------------------------------------------
__global__ __launch_bounds__(256)
void gemm_bf16_wmma(const float* __restrict__ A, const float* __restrict__ W,
                    const float* __restrict__ bias, float* __restrict__ Cout,
                    int M, int Nout, int K, int mode,
                    const float* __restrict__ wbuf, const float* __restrict__ attn_w,
                    float* __restrict__ partials, int Hdim, int nMask) {
    __shared__ __align__(16) unsigned char smem[33792];

    const int tid  = threadIdx.x;
    const int lane = tid & 31;
    const int wave = tid >> 5;        // 0..7
    const int waveM = wave >> 1;      // 0..3  (rows, 32 each)
    const int waveN = wave & 1;       // 0..1  (cols, 64 each)
    const int half = lane >> 4;       // 0/1
    const int lr   = lane & 15;       // 0..15

    const int rowBase = blockIdx.y * 128;
    const int colBase = blockIdx.x * 128;

    // staging: 256 threads move 128x32 fp32 per matrix per step (4 float4 each)
    const int ldRow = tid >> 3;       // 0..31  (+ i*32)
    const int ldKq  = tid & 7;        // float4 index within a 32-wide row

    v8f acc[2][4] = {};
    float4 ra[4], rb[4];

    // pointer helpers: single base cast + integer offsets (no pointer arrays!)
    auto sAp = [&](int buf) -> __bf16* {
        return (__bf16*)smem + buf * 4096;            // 4096 bf16 = 8KB
    };
    auto sBp = [&](int buf) -> __bf16* {
        return (__bf16*)smem + 8192 + buf * 4096;
    };

    auto loadRegs = [&](int k0) {
#pragma unroll
        for (int i = 0; i < 4; ++i) {
            int row = ldRow + i * 32;
            ra[i] = *(const float4*)(A + (size_t)(rowBase + row) * K + k0 + ldKq * 4);
            rb[i] = *(const float4*)(W + (size_t)(colBase + row) * K + k0 + ldKq * 4);
        }
    };
    auto cvtStore = [&](int buf) {
        __bf16* dA = sAp(buf);
        __bf16* dB = sBp(buf);
#pragma unroll
        for (int i = 0; i < 4; ++i) {
            int off = (ldRow + i * 32) * 32 + ldKq * 4;
            // native fp32->bf16 converts (v_cvt_pk), one 8B ds_store each
            v4bf pa = { (__bf16)ra[i].x, (__bf16)ra[i].y,
                        (__bf16)ra[i].z, (__bf16)ra[i].w };
            v4bf pb = { (__bf16)rb[i].x, (__bf16)rb[i].y,
                        (__bf16)rb[i].z, (__bf16)rb[i].w };
            *(v4bf*)(dA + off) = pa;
            *(v4bf*)(dB + off) = pb;
        }
    };
    auto loadFrag = [&](const __bf16* s, int rowOff) -> v16bf {
        union { v16bf v; v8bf h[2]; } u;
        int base = (rowOff + lr) * 32 + half * 8;
        u.h[0] = *(const v8bf*)(s + base);
        u.h[1] = *(const v8bf*)(s + base + 16);
        return u.v;
    };

    const int nsteps = K >> 5;
    loadRegs(0);
    cvtStore(0);
    __syncthreads();

    for (int step = 0; step < nsteps; ++step) {
        const int cur = step & 1;
        if (step + 1 < nsteps) loadRegs((step + 1) << 5);   // prefetch next tile

        v16bf af[2], bfr[4];
        {
            const __bf16* cA = sAp(cur);
            const __bf16* cB = sBp(cur);
#pragma unroll
            for (int i = 0; i < 2; ++i) af[i]  = loadFrag(cA, waveM * 32 + i * 16);
#pragma unroll
            for (int j = 0; j < 4; ++j) bfr[j] = loadFrag(cB, waveN * 64 + j * 16);
        }
#pragma unroll
        for (int i = 0; i < 2; ++i)
#pragma unroll
            for (int j = 0; j < 4; ++j)
                acc[i][j] = __builtin_amdgcn_wmma_f32_16x16x32_bf16(
                    false, af[i], false, bfr[j], (short)0, acc[i][j], false, false);

        if (step + 1 < nsteps) cvtStore(cur ^ 1);
        __syncthreads();
    }

    // C/D layout: lanes 0-15: col=lr, rows r..r+7; lanes 16-31: col=lr, rows r+8..15
    if (mode == 0) {
#pragma unroll
        for (int i = 0; i < 2; ++i)
#pragma unroll
            for (int j = 0; j < 4; ++j) {
                int col   = colBase + waveN * 64 + j * 16 + lr;
                int rbase = rowBase + waveM * 32 + i * 16 + half * 8;
                float b = bias ? bias[col] : 0.0f;
#pragma unroll
                for (int r = 0; r < 8; ++r)
                    Cout[(size_t)(rbase + r) * Nout + col] = acc[i][j][r] + b;
            }
    } else {
        float* sRed = (float*)(smem + 32768);   // 128 rows x 2 col-waves
        // tanh + attn_w dot, then deterministic row reduction
#pragma unroll
        for (int i = 0; i < 2; ++i) {
            float vsum[8];
#pragma unroll
            for (int r = 0; r < 8; ++r) vsum[r] = 0.0f;
#pragma unroll
            for (int j = 0; j < 4; ++j) {
                int gc = colBase + waveN * 64 + j * 16 + lr;
                float bb = bias[gc];
                float aw = attn_w[gc];
                int rbase = rowBase + waveM * 32 + i * 16 + half * 8;
#pragma unroll
                for (int r = 0; r < 8; ++r) {
                    int n = (rbase + r) & nMask;
                    vsum[r] += tanhf(acc[i][j][r] + bb +
                                     wbuf[(size_t)n * Hdim + gc]) * aw;
                }
            }
            // reduce across the 16 lanes of each half-group (cols)
#pragma unroll
            for (int r = 0; r < 8; ++r) {
                float v = vsum[r];
                v += __shfl_xor(v, 1, 32);
                v += __shfl_xor(v, 2, 32);
                v += __shfl_xor(v, 4, 32);
                v += __shfl_xor(v, 8, 32);
                if (lr == 0) {
                    int row = waveM * 32 + i * 16 + half * 8 + r;   // 0..127
                    sRed[row * 2 + waveN] = v;
                }
            }
        }
        __syncthreads();
        if (tid < 128) {
            float s = sRed[tid * 2] + sRed[tid * 2 + 1];
            partials[(size_t)(rowBase + tid) * gridDim.x + blockIdx.x] = s;
        }
    }
}

// ---------------------------------------------------------------------------
// energy[m] = attn_b + sum_j partials[m*nchunks + j]          (m = s*N + n)
// ---------------------------------------------------------------------------
__global__ void energy_reduce_kernel(const float* __restrict__ partials,
                                     const float* __restrict__ attn_b,
                                     float* __restrict__ energy, int nchunks) {
    int m = blockIdx.x * blockDim.x + threadIdx.x;
    if (m >= S_LEN * N_BATCH) return;
    float s = attn_b[0];
    for (int j = 0; j < nchunks; ++j) s += partials[(size_t)m * nchunks + j];
    energy[m] = s;
}

// ---------------------------------------------------------------------------
// softmax over the sequence dim (S=64) for each batch column n
// ---------------------------------------------------------------------------
__global__ void softmax_attn_kernel(const float* __restrict__ energy,
                                    float* __restrict__ att) {
    __shared__ float red[S_LEN];
    int n = blockIdx.x;
    int s = threadIdx.x;                       // 0..63
    float v = energy[s * N_BATCH + n];
    red[s] = v;
    __syncthreads();
    for (int off = 32; off > 0; off >>= 1) {
        if (s < off) red[s] = fmaxf(red[s], red[s + off]);
        __syncthreads();
    }
    float mx = red[0];
    __syncthreads();
    float e = expf(v - mx);
    red[s] = e;
    __syncthreads();
    for (int off = 32; off > 0; off >>= 1) {
        if (s < off) red[s] += red[s + off];
        __syncthreads();
    }
    att[s * N_BATCH + n] = e / red[0];
}

// ---------------------------------------------------------------------------
// context[n,d] = sum_s att[s,n] * enc[s,n,d]   -> rnn_input[n, 0:2048)
// ---------------------------------------------------------------------------
__global__ void context_kernel(const float* __restrict__ att,
                               const float* __restrict__ enc,
                               float* __restrict__ rnn_input) {
    __shared__ float aw[S_LEN];
    int n = blockIdx.y;
    int t = threadIdx.x;
    if (t < S_LEN) aw[t] = att[t * N_BATCH + n];
    __syncthreads();
    int d = blockIdx.x * 256 + t;              // 0..2047
    float acc = 0.0f;
#pragma unroll 8
    for (int s = 0; s < S_LEN; ++s)
        acc += aw[s] * enc[((size_t)s * N_BATCH + n) * DH + d];
    rnn_input[(size_t)n * RNN_IN + d] = acc;
}

// ---------------------------------------------------------------------------
// embedding lookup -> rnn_input[n, 2048:2304)
// ---------------------------------------------------------------------------
__global__ void embed_kernel(const int* __restrict__ x,
                             const float* __restrict__ emb,
                             float* __restrict__ rnn_input) {
    int n = blockIdx.x;
    int e = threadIdx.x;                       // 0..255
    rnn_input[(size_t)n * RNN_IN + DH + e] = emb[(size_t)x[n] * E_DIM + e];
}

// ---------------------------------------------------------------------------
// GRU gate combine (PyTorch cell math), writes h' to both destinations
// ---------------------------------------------------------------------------
__global__ void gru_combine_kernel(const float* __restrict__ gi,
                                   const float* __restrict__ gh,
                                   const float* __restrict__ hprev,
                                   float* __restrict__ hout_a,
                                   float* __restrict__ hout_b) {
    int idx = blockIdx.x * blockDim.x + threadIdx.x;   // N*H
    int n = idx >> 10;
    int c = idx & (H_DIM - 1);
    const float* gin = gi + (size_t)n * G3H;
    const float* ghn = gh + (size_t)n * G3H;
    float r  = sigmoidf_(gin[c] + ghn[c]);
    float z  = sigmoidf_(gin[H_DIM + c] + ghn[H_DIM + c]);
    float nn = tanhf(gin[2 * H_DIM + c] + r * ghn[2 * H_DIM + c]);
    float h  = (1.0f - z) * nn + z * hprev[idx];
    hout_a[idx] = h;
    hout_b[idx] = h;
}

// ---------------------------------------------------------------------------
extern "C" void kernel_launch(void* const* d_in, const int* in_sizes, int n_in,
                              void* d_out, int out_size, void* d_ws, size_t ws_size,
                              hipStream_t stream) {
    const int*   x      = (const int*)  d_in[0];
    const float* enc    = (const float*)d_in[1];
    const float* hidden = (const float*)d_in[2];
    // d_in[3] = cell (unused by reference math)
    const float* emb    = (const float*)d_in[4];
    const float* U_w    = (const float*)d_in[5];
    const float* U_b    = (const float*)d_in[6];
    const float* W_w    = (const float*)d_in[7];
    const float* W_b    = (const float*)d_in[8];
    const float* attn_w = (const float*)d_in[9];
    const float* attn_b = (const float*)d_in[10];
    const float* W_ih0  = (const float*)d_in[11];
    const float* W_hh0  = (const float*)d_in[12];
    const float* b_ih0  = (const float*)d_in[13];
    const float* b_hh0  = (const float*)d_in[14];
    const float* W_ih1  = (const float*)d_in[15];
    const float* W_hh1  = (const float*)d_in[16];
    const float* b_ih1  = (const float*)d_in[17];
    const float* b_hh1  = (const float*)d_in[18];
    const float* fc_w   = (const float*)d_in[19];
    const float* fc_b   = (const float*)d_in[20];

    const float* hidden0 = hidden;
    const float* hidden1 = hidden + (size_t)N_BATCH * H_DIM;

    float* out = (float*)d_out;
    float* out_pred = out;                                          // (N,V)      32768
    float* out_hid0 = out + (size_t)N_BATCH * V_DIM;                // (N,H)     262144
    float* out_hid1 = out_hid0 + (size_t)N_BATCH * H_DIM;           // (N,H)     262144
    float* out_att  = out_hid1 + (size_t)N_BATCH * H_DIM;           // (S,N,1)    16384

    float* ws = (float*)d_ws;
    float* wbuf     = ws;                   // N*H            = 262144
    float* partials = wbuf + 262144;        // 16384*8 (<=262144)
    float* energy   = partials + 262144;    // S*N            =  16384
    float* rnn      = energy + 16384;       // N*RNN_IN       = 589824
    float* gi       = rnn + 589824;         // N*3H           = 786432
    float* gh       = gi + 786432;          // N*3H           = 786432
    float* h0buf    = gh + 786432;          // N*H            = 262144
    float* h1buf    = h0buf + 262144;       // N*H            = 262144

    dim3 blk(256);

    // 1) w = hidden[1] @ W_w^T + W_b                     (256 x 1024, K=1024)
    gemm_bf16_wmma<<<dim3(H_DIM / 128, N_BATCH / 128), blk, 0, stream>>>(
        hidden1, W_w, W_b, wbuf, N_BATCH, H_DIM, H_DIM, 0,
        nullptr, nullptr, nullptr, 0, 0);

    // 2) fused u-GEMM + tanh + attn_w dot -> row partials (16384 x 1024, K=2048)
    gemm_bf16_wmma<<<dim3(H_DIM / 128, (S_LEN * N_BATCH) / 128), blk, 0, stream>>>(
        enc, U_w, U_b, nullptr, S_LEN * N_BATCH, H_DIM, DH, 1,
        wbuf, attn_w, partials, H_DIM, N_BATCH - 1);

    // 3) energy = attn_b + sum(partials)
    energy_reduce_kernel<<<(S_LEN * N_BATCH) / 256, 256, 0, stream>>>(
        partials, attn_b, energy, H_DIM / 128);

    // 4) softmax over S -> attention output
    softmax_attn_kernel<<<N_BATCH, S_LEN, 0, stream>>>(energy, out_att);

    // 5) context -> rnn_input[:, 0:2048)
    context_kernel<<<dim3(DH / 256, N_BATCH), 256, 0, stream>>>(out_att, enc, rnn);

    // 6) embedding -> rnn_input[:, 2048:2304)
    embed_kernel<<<N_BATCH, E_DIM, 0, stream>>>(x, emb, rnn);

    // 7) GRU layer 0 gates
    gemm_bf16_wmma<<<dim3(G3H / 128, N_BATCH / 128), blk, 0, stream>>>(
        rnn, W_ih0, b_ih0, gi, N_BATCH, G3H, RNN_IN, 0, nullptr, nullptr, nullptr, 0, 0);
    gemm_bf16_wmma<<<dim3(G3H / 128, N_BATCH / 128), blk, 0, stream>>>(
        hidden0, W_hh0, b_hh0, gh, N_BATCH, G3H, H_DIM, 0, nullptr, nullptr, nullptr, 0, 0);
    gru_combine_kernel<<<(N_BATCH * H_DIM) / 256, 256, 0, stream>>>(
        gi, gh, hidden0, out_hid0, h0buf);

    // 8) GRU layer 1 gates
    gemm_bf16_wmma<<<dim3(G3H / 128, N_BATCH / 128), blk, 0, stream>>>(
        h0buf, W_ih1, b_ih1, gi, N_BATCH, G3H, H_DIM, 0, nullptr, nullptr, nullptr, 0, 0);
    gemm_bf16_wmma<<<dim3(G3H / 128, N_BATCH / 128), blk, 0, stream>>>(
        hidden1, W_hh1, b_hh1, gh, N_BATCH, G3H, H_DIM, 0, nullptr, nullptr, nullptr, 0, 0);
    gru_combine_kernel<<<(N_BATCH * H_DIM) / 256, 256, 0, stream>>>(
        gi, gh, hidden1, out_hid1, h1buf);

    // 9) predictions = h1 @ fc_w^T + fc_b               (256 x 128, K=1024)
    gemm_bf16_wmma<<<dim3(V_DIM / 128, N_BATCH / 128), blk, 0, stream>>>(
        h1buf, fc_w, fc_b, out_pred, N_BATCH, V_DIM, H_DIM, 0,
        nullptr, nullptr, nullptr, 0, 0);
}
